// Cheby_KAN_25838523252878
// MI455X (gfx1250) — matrix operand order repaired and checked by reading, not compiled, not run on hardware
//
#include <hip/hip_runtime.h>
#include <stdint.h>

typedef __attribute__((ext_vector_type(16))) __bf16 v16bf;
typedef __attribute__((ext_vector_type(8)))  float  v8f;

#define BLOCK_THREADS 256
#define MT     128           // batch rows per block
#define PCH    64            // p-features per basis chunk
#define NTRM   9             // deg+1
#define KCH    (PCH*NTRM)    // 576, K per chunk (multiple of 32)
#define MID    128
#define IN_DIM 8
#define NFEAT  32
#define K4     (MID*NTRM)    // 1152

union Frag32B { uint4 q[2]; v16bf v; };

__device__ __forceinline__ unsigned short f2bf(float f){
  union { float f; uint32_t u; } c; c.f = f;
  uint32_t u = c.u;
  return (unsigned short)((u + 0x7FFFu + ((u >> 16) & 1u)) >> 16); // RNE
}
__device__ __forceinline__ float bf2f(unsigned short h){
  union { uint32_t u; float f; } c; c.u = ((uint32_t)h) << 16;
  return c.f;
}

// Fold Wc[q][k] = tW[q][p][n] * W[q][p], bf16, with chunked n-major K-ordering:
//   k = ch*KCH + n*PCH + pl,  p = ch*PCH + pl
__global__ void prep_weights_kernel(const float* __restrict__ W,
                                    const float* __restrict__ tW,
                                    unsigned short* __restrict__ Wc,
                                    int q, int P){
  int K = P * NTRM;
  int idx = blockIdx.x * blockDim.x + threadIdx.x;
  if (idx >= q * K) return;
  int qq   = idx / K;
  int kidx = idx % K;
  int ch  = kidx / KCH;
  int rem = kidx % KCH;
  int n   = rem / PCH;
  int pl  = rem % PCH;
  int pp  = ch * PCH + pl;
  Wc[idx] = f2bf(tW[(qq * P + pp) * NTRM + n] * W[qq * P + pp]);
}

__global__ __launch_bounds__(BLOCK_THREADS)
void kan_fused_kernel(const float* __restrict__ x,
                      const float* __restrict__ B,
                      const unsigned short* __restrict__ Wc0,
                      const unsigned short* __restrict__ Wc1,
                      const unsigned short* __restrict__ Wc2,
                      const unsigned short* __restrict__ Wc3,
                      const unsigned short* __restrict__ Wc4,
                      float* __restrict__ out, int batch)
{
  extern __shared__ char smem_raw[];
  unsigned short (*basis)[KCH] = (unsigned short (*)[KCH])(smem_raw);                 // 147456 B
  float (*hbuf)[MID]           = (float (*)[MID])(smem_raw + (size_t)MT*KCH*2);       //  65536 B
  float (*xs)[IN_DIM]          = (float (*)[IN_DIM])(smem_raw + (size_t)MT*KCH*2 + (size_t)MT*MID*4);
  float (*Bs)[NFEAT]           = (float (*)[NFEAT])((char*)xs + (size_t)MT*IN_DIM*4);
  unsigned short* wc4s         = (unsigned short*)((char*)Bs + (size_t)IN_DIM*NFEAT*4);

  const int t     = threadIdx.x;
  const int lane  = t & 31;
  const int wave  = t >> 5;        // 0..7 : this wave's 16-col N-tile
  const int lrow  = lane & 15;
  const int khalf = lane >> 4;
  const int row0  = blockIdx.x * MT;

  // ---- stage x tile, B, final-layer weights ----
  for (int i = t; i < MT*IN_DIM; i += BLOCK_THREADS){
    int gi = row0*IN_DIM + i;
    ((float*)xs)[i] = (gi < batch*IN_DIM) ? x[gi] : 0.f;
  }
  for (int i = t; i < IN_DIM*NFEAT; i += BLOCK_THREADS) ((float*)Bs)[i] = B[i];
  for (int i = t; i < K4; i += BLOCK_THREADS)           wc4s[i] = Wc4[i];
  __syncthreads();

  // ---- Fourier features: hbuf[r][0..63] = [sin|cos](2*pi*x@B) ----
  for (int i = t; i < MT*64; i += BLOCK_THREADS){
    int r = i >> 6, j = i & 63, jj = j & 31;
    float s = 0.f;
    #pragma unroll
    for (int c = 0; c < IN_DIM; ++c) s += xs[r][c] * Bs[c][jj];
    float ph = 6.283185307179586f * s;
    hbuf[r][j] = (j < NFEAT) ? sinf(ph) : cosf(ph);
  }

  // basis chunk ch, n-major: basis[r][n*PCH + pl] = T_n(tanh(hbuf[r][ch*PCH+pl]))
  // each thread processes a pair of adjacent p's -> packed 32-bit LDS stores.
  auto build_basis = [&](int ch){
    for (int i = t; i < MT*PCH/2; i += BLOCK_THREADS){
      int r  = i >> 5;               // 32 p-pairs per row
      int pl = (i & 31) * 2;
      float xt0 = tanhf(hbuf[r][ch*PCH + pl]);
      float xt1 = tanhf(hbuf[r][ch*PCH + pl + 1]);
      uint32_t* bp = (uint32_t*)&basis[r][pl];   // stride PCH shorts per n (= PCH/2 u32)
      bp[0] = 0x3F803F80u;                       // T0 = 1.0, 1.0
      float a2 = 1.f, a1 = 2.f*xt0;              // T1 = 2*xt (KAN variant)
      float b2 = 1.f, b1 = 2.f*xt1;
      bp[PCH/2] = (uint32_t)f2bf(a1) | ((uint32_t)f2bf(b1) << 16);
      #pragma unroll
      for (int n = 2; n < NTRM; ++n){
        float an = 2.f*xt0*a1 - a2;  a2 = a1;  a1 = an;
        float bn = 2.f*xt1*b1 - b2;  b2 = b1;  b1 = bn;
        bp[n*(PCH/2)] = (uint32_t)f2bf(an) | ((uint32_t)f2bf(bn) << 16);
      }
    }
  };

  // one KAN layer: hbuf[:, :128] = basis @ Wc^T (+ residual), software-pipelined frags
  auto run_layer = [&](const unsigned short* __restrict__ Wc, int nch, bool residual){
    v8f acc[8] = {};
    const int K = nch * KCH;
    for (int ch = 0; ch < nch; ++ch){
      __syncthreads();                 // prior consumers of basis / producers of hbuf done
      build_basis(ch);
      __syncthreads();
      // B fragment column: N = wave*16 + lrow, K-half by lane
      const unsigned short* bcol = Wc + (size_t)(wave*16 + lrow) * K + ch*KCH + khalf*8;
      Frag32B bfc, bfn;
      bfc.q[0] = ((const uint4*)(bcol))[0];
      bfc.q[1] = ((const uint4*)(bcol))[2];
      for (int k0 = 0; k0 < KCH; k0 += 32){
        if (k0 + 32 < KCH){            // prefetch next k-step's B fragment
          bfn.q[0] = ((const uint4*)(bcol + k0 + 32))[0];
          bfn.q[1] = ((const uint4*)(bcol + k0 + 32))[2];
        }
        Frag32B afc, afn;
        {
          const unsigned short* a0 = &basis[lrow][k0 + khalf*8];
          afc.q[0] = *(const uint4*)(a0);
          afc.q[1] = *(const uint4*)(a0 + 16);
        }
        #pragma unroll
        for (int mt = 0; mt < 8; ++mt){
          if (mt < 7){                 // prefetch next m-tile's A fragment
            const unsigned short* an = &basis[(mt+1)*16 + lrow][k0 + khalf*8];
            afn.q[0] = *(const uint4*)(an);
            afn.q[1] = *(const uint4*)(an + 16);
          }
          acc[mt] = __builtin_amdgcn_wmma_f32_16x16x32_bf16(
              false, afc.v, false, bfc.v, (short)0, acc[mt], false, false);
          if (mt < 7) afc = afn;
        }
        if (k0 + 32 < KCH) bfc = bfn;
      }
    }
    __syncthreads();
    // C/D layout: VGPR v -> M = v + 8*khalf (per 16-row tile), N = lrow
    const int col = wave*16 + lrow;
    #pragma unroll
    for (int mt = 0; mt < 8; ++mt){
      #pragma unroll
      for (int v = 0; v < 8; ++v){
        int m = mt*16 + khalf*8 + v;
        float val = acc[mt][v];
        if (residual) val += hbuf[m][col];
        hbuf[m][col] = val;
      }
    }
  };

  run_layer(Wc0, 1, false);
  run_layer(Wc1, 2, true);
  run_layer(Wc2, 2, true);
  run_layer(Wc3, 2, true);

  // ---- final layer (q = 1) + Chebyshev-GeLU head ----
  float s = 0.f;
  for (int ch = 0; ch < 2; ++ch){
    __syncthreads();
    build_basis(ch);
    __syncthreads();
    if (t < MT){
      const unsigned short* bp = &basis[t][0];
      const unsigned short* wp = &wc4s[ch*KCH];
      float a = 0.f;
      for (int k = 0; k < KCH; ++k) a += bf2f(bp[k]) * bf2f(wp[k]);
      s += a;
    }
  }
  if (t < MT && (row0 + t) < batch){
    float xt = tanhf(s);                       // standard T1 = xt here
    const float inv_sqrt2 = 0.7071067811865476f;
    float tm2 = 1.f, tm1 = xt;
    float g  = 0.5f*tm2*(1.f + erff(tm2*inv_sqrt2));
    g       += 0.5f*tm1*(1.f + erff(tm1*inv_sqrt2));
    #pragma unroll
    for (int n = 2; n <= 5; ++n){
      float tn = 2.f*xt*tm1 - tm2;
      g += 0.5f*tn*(1.f + erff(tn*inv_sqrt2));
      tm2 = tm1; tm1 = tn;
    }
    out[row0 + t] = g;
  }
}

extern "C" void kernel_launch(void* const* d_in, const int* in_sizes, int n_in,
                              void* d_out, int out_size, void* d_ws, size_t ws_size,
                              hipStream_t stream){
  const float* x = (const float*)d_in[0];
  const float* B = (const float*)d_in[1];
  const float* W[5]  = {(const float*)d_in[2], (const float*)d_in[4], (const float*)d_in[6],
                        (const float*)d_in[8], (const float*)d_in[10]};
  const float* tW[5] = {(const float*)d_in[3], (const float*)d_in[5], (const float*)d_in[7],
                        (const float*)d_in[9], (const float*)d_in[11]};
  const int qdim[5] = {128,128,128,128,1};
  const int pdim[5] = { 64,128,128,128,128};

  // carve workspace for folded bf16 weights (256B-aligned slices)
  unsigned short* wc[5];
  size_t off = 0;
  char* ws = (char*)d_ws;
  for (int l = 0; l < 5; ++l){
    wc[l] = (unsigned short*)(ws + off);
    size_t bytes = (size_t)qdim[l]*pdim[l]*NTRM*sizeof(unsigned short);
    off += (bytes + 255) & ~(size_t)255;
  }
  for (int l = 0; l < 5; ++l){
    int total = qdim[l]*pdim[l]*NTRM;
    prep_weights_kernel<<<(total + 255)/256, 256, 0, stream>>>(W[l], tW[l], wc[l], qdim[l], pdim[l]);
  }

  int batch = in_sizes[0] / IN_DIM;
  int grid  = (batch + MT - 1) / MT;
  size_t smem = (size_t)MT*KCH*2      // basis
              + (size_t)MT*MID*4      // hbuf
              + (size_t)MT*IN_DIM*4   // xs
              + (size_t)IN_DIM*NFEAT*4// Bs
              + (size_t)K4*2;         // wc4s
  kan_fused_kernel<<<grid, BLOCK_THREADS, smem, stream>>>(
      x, B, wc[0], wc[1], wc[2], wc[3], wc[4], (float*)d_out, batch);
}
